// StoichNet_73083163509423
// MI455X (gfx1250) — matrix operand order repaired and checked by reading, not compiled, not run on hardware
//
#include <hip/hip_runtime.h>
#include <hip/hip_bf16.h>

// ---------------------------------------------------------------------------
// Types for CDNA5 WMMA (gfx1250, wave32)
// ---------------------------------------------------------------------------
typedef __attribute__((ext_vector_type(16))) __bf16 v16bf;
typedef __attribute__((ext_vector_type(8)))  float  v8f;

// ---------------------------------------------------------------------------
// Network dims
// ---------------------------------------------------------------------------
#define ELEM_DIM 200
#define REACT_DIM 128
#define IN_DIM 328            // 200 + 128
#define IN_DIM_PAD 352        // padded to multiple of 32
#define N_HEADS 3

// Workspace layout: bf16 weights [H][O][Kp] (transposed, K padded), then fp32.
constexpr int OFF_W0 = 0;
constexpr int OFF_R0 = OFF_W0 + 3 * 256 * 352;
constexpr int OFF_W1 = OFF_R0 + 3 * 256 * 352;
constexpr int OFF_W2 = OFF_W1 + 3 * 256 * 256;
constexpr int OFF_W3 = OFF_W2 + 3 * 256 * 256;
constexpr int OFF_W4 = OFF_W3 + 3 * 256 * 256;
constexpr int OFF_R4 = OFF_W4 + 3 * 128 * 256;
constexpr int OFF_W5 = OFF_R4 + 3 * 128 * 256;
constexpr int OFF_W6 = OFF_W5 + 3 * 128 * 128;
constexpr int OFF_R6 = OFF_W6 + 3 * 64 * 128;
constexpr int W_TOTAL = OFF_R6 + 3 * 64 * 128;      // bf16 elements
constexpr size_t W_BYTES = (size_t)W_TOTAL * 2;     // 2,850,816 B

// Monotone float<->int mapping for atomicMax on floats (involution).
__device__ __forceinline__ int flipf(float f) {
  int i = __float_as_int(f);
  return i < 0 ? (i ^ 0x7fffffff) : i;
}
__device__ __forceinline__ float unflipf(int i) {
  int j = i < 0 ? (i ^ 0x7fffffff) : i;
  return __int_as_float(j);
}

__device__ __forceinline__ v8f wmma_bf16(v16bf a, v16bf b, v8f c) {
  return __builtin_amdgcn_wmma_f32_16x16x32_bf16(false, a, false, b, (short)0,
                                                 c, false, false);
}

// ---------------------------------------------------------------------------
// Weight prep: fp32 [H][K][O] -> bf16 [H][O][Kp] (transposed, K zero-padded)
// ---------------------------------------------------------------------------
__global__ void k_prep_w(const float* __restrict__ src, __bf16* __restrict__ dst,
                         int K, int Kp, int O, int total) {
  int i = blockIdx.x * blockDim.x + threadIdx.x;
  if (i >= total) return;
  int k = i % Kp;
  int t = i / Kp;
  int o = t % O;
  int h = t / O;
  float v = (k < K) ? src[((size_t)h * K + k) * O + o] : 0.f;
  dst[i] = (__bf16)v;
}

__global__ void k_seg_init(int* __restrict__ smax, float* __restrict__ ssum, int total) {
  int i = blockIdx.x * blockDim.x + threadIdx.x;
  if (i >= total) return;
  smax[i] = (int)0x807FFFFF;  // flipf(-inf)
  ssum[i] = 0.f;
}

// ---------------------------------------------------------------------------
// One residual layer, fully specialized at compile time.
// Each wave owns 16 rows and computes TWO 16-wide column tiles per pass
// (A-fragment reused across two B-fragments, two independent WMMA chains).
// ---------------------------------------------------------------------------
template <int KSTEPS, int KP_, int O_, int SI_, int SO_, bool PROJ>
__device__ __forceinline__ void run_layer(const __bf16* __restrict__ in,
                                          __bf16* __restrict__ out,
                                          const __bf16* __restrict__ Wt,
                                          const __bf16* __restrict__ Rt,
                                          const float* __restrict__ bh,
                                          int mrow, int nlane, int kbA, int kbB,
                                          int crow) {
  const __bf16* ar0 = in + (mrow + nlane) * SI_ + kbA;

#pragma unroll 1
  for (int ct = 0; ct < O_ / 32; ++ct) {
    const int n0 = ct * 32;  // two 16-wide output tiles: n0 and n0+16
    v8f acc0 = {}, acc1 = {};
    v8f accR0 = {}, accR1 = {};
    const __bf16* wr0 = Wt + (size_t)(n0 + nlane) * KP_ + kbB;
    const __bf16* wr1 = wr0 + 16 * KP_;
    const __bf16* rr0 = PROJ ? (Rt + (size_t)(n0 + nlane) * KP_ + kbB) : nullptr;
    const __bf16* rr1 = PROJ ? (rr0 + 16 * KP_) : nullptr;

#pragma unroll
    for (int ks = 0; ks < KSTEPS; ++ks) {
      const int k0 = ks * 32;
      v16bf a, b0, b1;
      // A fragment: two 16B LDS chunks (K = kbA+0..7 and kbA+8..15)
      ((uint4*)&a)[0] = *(const uint4*)(ar0 + k0);
      ((uint4*)&a)[1] = *(const uint4*)(ar0 + k0 + 8);
      // B fragments: 16 contiguous bf16 per lane of transposed weights (L2-hot)
      ((uint4*)&b0)[0] = *(const uint4*)(wr0 + k0);
      ((uint4*)&b0)[1] = *(const uint4*)(wr0 + k0 + 8);
      ((uint4*)&b1)[0] = *(const uint4*)(wr1 + k0);
      ((uint4*)&b1)[1] = *(const uint4*)(wr1 + k0 + 8);
      acc0 = wmma_bf16(a, b0, acc0);
      acc1 = wmma_bf16(a, b1, acc1);
      if (PROJ) {
        v16bf r0v, r1v;
        ((uint4*)&r0v)[0] = *(const uint4*)(rr0 + k0);
        ((uint4*)&r0v)[1] = *(const uint4*)(rr0 + k0 + 8);
        ((uint4*)&r1v)[0] = *(const uint4*)(rr1 + k0);
        ((uint4*)&r1v)[1] = *(const uint4*)(rr1 + k0 + 8);
        accR0 = wmma_bf16(a, r0v, accR0);
        accR1 = wmma_bf16(a, r1v, accR1);
      }
    }

    // bias + ReLU + residual (projection or identity), store bf16
    const float bn0 = bh[n0 + nlane];
    const float bn1 = bh[n0 + 16 + nlane];
#pragma unroll
    for (int r = 0; r < 8; ++r) {
      const int m = mrow + crow + r;
      float v0 = fmaxf(acc0[r] + bn0, 0.f);
      float v1 = fmaxf(acc1[r] + bn1, 0.f);
      float res0, res1;
      if (PROJ) {
        res0 = accR0[r];
        res1 = accR1[r];
      } else {
        res0 = (float)in[m * SI_ + n0 + nlane];
        res1 = (float)in[m * SI_ + n0 + 16 + nlane];
      }
      out[m * SO_ + n0 + nlane] = (__bf16)(v0 + res0);
      out[m * SO_ + n0 + 16 + nlane] = (__bf16)(v1 + res1);
    }
  }
}

// ---------------------------------------------------------------------------
// Fused per-head residual MLP: 64 rows per block, all activations in LDS.
// ---------------------------------------------------------------------------
__global__ __launch_bounds__(128) void k_fused_mlp(
    const float* __restrict__ elem, const int* __restrict__ ridx,
    const float* __restrict__ remb, const __bf16* __restrict__ w,
    const float* __restrict__ b0, const float* __restrict__ b1,
    const float* __restrict__ b2, const float* __restrict__ b3,
    const float* __restrict__ b4, const float* __restrict__ b5,
    const float* __restrict__ b6,
    const float* __restrict__ outW, const float* __restrict__ outb,
    float* __restrict__ gate, int* __restrict__ smax, int nE, int C) {
  // Ping-pong activation tiles (bf16), strides padded for 16B-aligned b128
  // LDS reads with bank spread: 360/264/136/72 elements.
  __shared__ __align__(16) __bf16 sA[64 * 360];
  __shared__ __align__(16) __bf16 sB[64 * 264];

  const int head = blockIdx.y;
  const int row0 = blockIdx.x * 64;
  const int tid = threadIdx.x;

  // ---- gather + concat + pad layer-0 input tile, fp32 -> bf16 ----
  for (int e = tid; e < 64 * IN_DIM_PAD; e += 128) {
    const int r = e / IN_DIM_PAD;
    const int c = e - r * IN_DIM_PAD;
    const int gr = row0 + r;
    float v = 0.f;
    if (gr < nE) {
      if (c < ELEM_DIM) v = elem[(size_t)gr * ELEM_DIM + c];
      else if (c < IN_DIM) v = remb[(size_t)ridx[gr] * REACT_DIM + (c - ELEM_DIM)];
    }
    sA[r * 360 + c] = (__bf16)v;
  }
  __syncthreads();

  const int lane = tid & 31;
  const int wave = tid >> 5;        // 0..3 -> 16 rows each
  const int nlane = lane & 15;      // N column within 16-wide tile / A row
  const bool hi = lane >= 16;
  const int mrow = wave * 16;
  const int kbA = hi ? 16 : 0;      // A frag K base (contiguous k = 16h + s)
  const int kbB = hi ? 16 : 0;      // B frag K base (contiguous k = 16h + s, same as A)
  const int crow = hi ? 8 : 0;      // C/D frag row base

  // 7 specialized layers; ping-pong sA <-> sB. No inter-wave barriers needed:
  // each wave reads/writes only its own 16 rows.
  run_layer<11, 352, 256, 360, 264, true>(sA, sB, w + OFF_W0 + head * 256 * 352,
                                          w + OFF_R0 + head * 256 * 352,
                                          b0 + head * 256, mrow, nlane, kbA, kbB, crow);
  run_layer<8, 256, 256, 264, 264, false>(sB, sA, w + OFF_W1 + head * 256 * 256,
                                          nullptr, b1 + head * 256, mrow, nlane,
                                          kbA, kbB, crow);
  run_layer<8, 256, 256, 264, 264, false>(sA, sB, w + OFF_W2 + head * 256 * 256,
                                          nullptr, b2 + head * 256, mrow, nlane,
                                          kbA, kbB, crow);
  run_layer<8, 256, 256, 264, 264, false>(sB, sA, w + OFF_W3 + head * 256 * 256,
                                          nullptr, b3 + head * 256, mrow, nlane,
                                          kbA, kbB, crow);
  run_layer<8, 256, 128, 264, 136, true>(sA, sB, w + OFF_W4 + head * 128 * 256,
                                         w + OFF_R4 + head * 128 * 256,
                                         b4 + head * 128, mrow, nlane, kbA, kbB, crow);
  run_layer<4, 128, 128, 136, 136, false>(sB, sA, w + OFF_W5 + head * 128 * 128,
                                          nullptr, b5 + head * 128, mrow, nlane,
                                          kbA, kbB, crow);
  run_layer<4, 128, 64, 136, 72, true>(sA, sB, w + OFF_W6 + head * 64 * 128,
                                       w + OFF_R6 + head * 64 * 128,
                                       b6 + head * 64, mrow, nlane, kbA, kbB, crow);
  __syncthreads();

  // ---- output projection (64 -> 1), gate store, segment max ----
  if (tid < 64) {
    const int gr = row0 + tid;
    if (gr < nE) {
      const float* ow = outW + head * 64;
      float g = outb[head];
      const __bf16* hrow = sB + tid * 72;
#pragma unroll 8
      for (int k = 0; k < 64; ++k) g += (float)hrow[k] * ow[k];
      gate[(size_t)head * nE + gr] = g;
      atomicMax(&smax[head * C + ridx[gr]], flipf(g));
    }
  }
}

// ---------------------------------------------------------------------------
// Segment softmax passes
// ---------------------------------------------------------------------------
__global__ void k_exp(float* __restrict__ gate, const int* __restrict__ ridx,
                      const int* __restrict__ smax, float* __restrict__ ssum,
                      int nE, int C) {
  int i = blockIdx.x * blockDim.x + threadIdx.x;
  if (i >= 3 * nE) return;
  int h = i / nE;
  int n = i - h * nE;
  int seg = ridx[n];
  float m = unflipf(smax[h * C + seg]);
  float e = __expf(gate[i] - m);
  gate[i] = e;
  atomicAdd(&ssum[h * C + seg], e);
}

__global__ void k_final(const float* __restrict__ gate, const int* __restrict__ ridx,
                        const float* __restrict__ ssum, float* __restrict__ out,
                        int nE, int C) {
  int n = blockIdx.x * blockDim.x + threadIdx.x;
  if (n >= nE) return;
  int seg = ridx[n];
  float acc = 0.f;
#pragma unroll
  for (int h = 0; h < N_HEADS; ++h)
    acc += gate[(size_t)h * nE + n] / (ssum[h * C + seg] + 1e-13f);
  out[n] = acc * (1.f / 3.f);
}

// ---------------------------------------------------------------------------
// Launch
// ---------------------------------------------------------------------------
extern "C" void kernel_launch(void* const* d_in, const int* in_sizes, int n_in,
                              void* d_out, int out_size, void* d_ws, size_t ws_size,
                              hipStream_t stream) {
  const float* elem = (const float*)d_in[0];
  const int*   ridx = (const int*)d_in[1];
  const float* remb = (const float*)d_in[2];
  const float* fcW0 = (const float*)d_in[3];
  const float* fcb0 = (const float*)d_in[4];
  const float* fcW1 = (const float*)d_in[5];
  const float* fcb1 = (const float*)d_in[6];
  const float* fcW2 = (const float*)d_in[7];
  const float* fcb2 = (const float*)d_in[8];
  const float* fcW3 = (const float*)d_in[9];
  const float* fcb3 = (const float*)d_in[10];
  const float* fcW4 = (const float*)d_in[11];
  const float* fcb4 = (const float*)d_in[12];
  const float* fcW5 = (const float*)d_in[13];
  const float* fcb5 = (const float*)d_in[14];
  const float* fcW6 = (const float*)d_in[15];
  const float* fcb6 = (const float*)d_in[16];
  const float* resW0 = (const float*)d_in[17];
  const float* resW4 = (const float*)d_in[18];
  const float* resW6 = (const float*)d_in[19];
  const float* outW = (const float*)d_in[20];
  const float* outb = (const float*)d_in[21];

  const int nE = in_sizes[0] / ELEM_DIM;   // 200000
  const int C  = in_sizes[2] / REACT_DIM;  // 100000

  __bf16* wbf = (__bf16*)d_ws;
  char* wsb = (char*)d_ws;
  float* gate = (float*)(wsb + W_BYTES);
  int*   smax = (int*)(wsb + W_BYTES + (size_t)3 * nE * 4);
  float* ssum = (float*)(wsb + W_BYTES + (size_t)3 * nE * 4 + (size_t)3 * C * 4);

  // 1) weights -> bf16, transposed + K-padded
#define PREP(src, off, K, Kp, O)                                                 \
  do {                                                                           \
    int tot = 3 * (O) * (Kp);                                                    \
    k_prep_w<<<(tot + 255) / 256, 256, 0, stream>>>(src, wbf + (off), K, Kp, O,  \
                                                    tot);                        \
  } while (0)
  PREP(fcW0, OFF_W0, 328, 352, 256);
  PREP(resW0, OFF_R0, 328, 352, 256);
  PREP(fcW1, OFF_W1, 256, 256, 256);
  PREP(fcW2, OFF_W2, 256, 256, 256);
  PREP(fcW3, OFF_W3, 256, 256, 256);
  PREP(fcW4, OFF_W4, 256, 256, 128);
  PREP(resW4, OFF_R4, 256, 256, 128);
  PREP(fcW5, OFF_W5, 128, 128, 128);
  PREP(fcW6, OFF_W6, 128, 128, 64);
  PREP(resW6, OFF_R6, 128, 128, 64);
#undef PREP

  // 2) segment reduction init
  {
    int tot = 3 * C;
    k_seg_init<<<(tot + 255) / 256, 256, 0, stream>>>(smax, ssum, tot);
  }

  // 3) fused gather + 7-layer residual MLP + gate + segment max
  {
    dim3 grid((nE + 63) / 64, N_HEADS);
    k_fused_mlp<<<grid, 128, 0, stream>>>(elem, ridx, remb, wbf,
                                          fcb0, fcb1, fcb2, fcb3, fcb4, fcb5, fcb6,
                                          outW, outb, gate, smax, nE, C);
  }

  // 4) exp + segment sum
  {
    int tot = 3 * nE;
    k_exp<<<(tot + 255) / 256, 256, 0, stream>>>(gate, ridx, smax, ssum, nE, C);
  }

  // 5) normalize + mean over heads
  k_final<<<(nE + 255) / 256, 256, 0, stream>>>(gate, ridx, ssum, (float*)d_out,
                                                nE, C);
}
